// MPNN_74637941670412
// MI455X (gfx1250) — matrix operand order repaired and checked by reading, not compiled, not run on hardware
//
#include <hip/hip_runtime.h>
#include <hip/hip_bf16.h>
#include <math.h>

// ---------------------------------------------------------------------------
// ProteinMPNN-style encoder layer for MI455X (gfx1250, wave32, WMMA).
// L=8192 nodes, K=30 neighbors, H=128. All GEMMs in bf16 WMMA (f32 accum).
// ---------------------------------------------------------------------------

#define L_NODES 8192
#define K_NB    30
#define H_DIM   128
#define D_IN    384     // 3*H
#define F_DIM   512     // 4*H
#define INV_SCALE (1.0f/30.0f)
#define LN_EPS  1e-5f

typedef __attribute__((ext_vector_type(16))) __bf16 v16bf;
typedef __attribute__((ext_vector_type(8)))  float  v8f;

struct __align__(16) bf16x8 { __bf16 v[8]; };

// ---- workspace layout (bytes) ---------------------------------------------
#define OFF_W1T   0u          // [128][384] bf16
#define OFF_W2T   98304u      // [128][128] bf16
#define OFF_W3T   131072u
#define OFF_W11T  163840u
#define OFF_W12T  262144u
#define OFF_W13T  294912u
#define OFF_WIT   327680u     // [512][128] bf16
#define OFF_WOT   458752u     // [128][512] bf16
#define OFF_HV1   589824u     // [8192][128] f32

__device__ __forceinline__ float gelu_exact(float x) {
    return 0.5f * x * (1.0f + erff(x * 0.7071067811865476f));
}

// A-fragment for v_wmma_f32_16x16x32_bf16.
// Contiguous K order (matches B): lanes 0-15 (hi=0) hold K = 0..15;
// lanes 16-31 (hi=1) hold K = 16..31 of row M = lane&15.
__device__ __forceinline__ v16bf load_afrag(const __bf16* row, int kbase, int hi) {
    const __bf16* p = row + kbase + (hi ? 16 : 0);
    v16bf a;
#pragma unroll
    for (int j = 0; j < 16; ++j) a[j] = p[j];
    return a;
}

// B-fragment: lane<16 holds column N=lane&15, K=kbase..+15; lane>=16 K=kbase+16..+31.
// colptr points at Wt[col*din] (transposed weight row = contiguous K).
__device__ __forceinline__ v16bf load_bfrag(const __bf16* colptr, int kbase, int hi) {
    const __bf16* p = colptr + kbase + (hi ? 16 : 0);
    v16bf b;
#pragma unroll
    for (int j = 0; j < 16; ++j) b[j] = p[j];
    return b;
}

__device__ __forceinline__ v8f wmma_bf16(v16bf a, v16bf b, v8f c) {
    return __builtin_amdgcn_wmma_f32_16x16x32_bf16(false, a, false, b, (short)0, c,
                                                   false, false);
}

// Stage the [32,384] concat tile [hV_i | hE_ik | hV_j] as bf16 into LDS.
// 8-element chunks: two float4 global loads -> one 16B LDS store.
__device__ __forceinline__ void stage_concat_tile(
    const float* __restrict__ hVsrc, const float* __restrict__ hE,
    const int* __restrict__ Eidx, int i, int tid, __bf16 (*sEV)[D_IN]) {
    for (int cidx = tid; cidx < 32 * (D_IN / 8); cidx += 512) {
        int r = cidx / (D_IN / 8);
        int cc = (cidx - r * (D_IN / 8)) * 8;
        bf16x8 out;
        if (r < K_NB) {
            const float* src;
            if (cc < H_DIM)           src = hVsrc + (size_t)i * H_DIM + cc;
            else if (cc < 2 * H_DIM)  src = hE + (size_t)(i * K_NB + r) * H_DIM + (cc - H_DIM);
            else { int j = Eidx[i * K_NB + r];
                   src = hVsrc + (size_t)j * H_DIM + (cc - 2 * H_DIM); }
            float4 a = ((const float4*)src)[0];
            float4 b = ((const float4*)src)[1];
            out.v[0] = (__bf16)a.x; out.v[1] = (__bf16)a.y;
            out.v[2] = (__bf16)a.z; out.v[3] = (__bf16)a.w;
            out.v[4] = (__bf16)b.x; out.v[5] = (__bf16)b.y;
            out.v[6] = (__bf16)b.z; out.v[7] = (__bf16)b.w;
        } else {
#pragma unroll
            for (int j = 0; j < 8; ++j) out.v[j] = (__bf16)0.0f;
        }
        *(bf16x8*)&sEV[r][cc] = out;
    }
}

// ---------------------------------------------------------------------------
// Weight convert + transpose: src[din][dout] f32 -> dst[dout][din] bf16
// ---------------------------------------------------------------------------
__global__ void cvt_t_kernel(const float* __restrict__ src, __bf16* __restrict__ dst,
                             int din, int dout) {
    int idx = blockIdx.x * blockDim.x + threadIdx.x;
    if (idx < din * dout) {
        int k = idx / dout, o = idx - k * dout;
        dst[o * din + k] = (__bf16)src[idx];
    }
}

// ---------------------------------------------------------------------------
// Stage 1: node message MLP + masked K-sum + residual LayerNorm -> hV1 (ws)
// One block per node. 512 threads = 16 waves: wave -> (mt = wave&1, nt = wave>>1).
// ---------------------------------------------------------------------------
__global__ __launch_bounds__(512) void node_msg_kernel(
    const float* __restrict__ hV, const float* __restrict__ hE,
    const int* __restrict__ Eidx, const float* __restrict__ mask_att,
    const float* __restrict__ b1, const float* __restrict__ b2,
    const float* __restrict__ b3,
    const __bf16* __restrict__ W1t, const __bf16* __restrict__ W2t,
    const __bf16* __restrict__ W3t,
    const float* __restrict__ g1, const float* __restrict__ be1,
    float* __restrict__ hV1) {
    const int i = blockIdx.x;
    const int tid = threadIdx.x;
    const int lane = tid & 31, wave = tid >> 5;
    const int hi = lane >> 4, ln = lane & 15;
    const int mt = wave & 1, nt = wave >> 1;

    __shared__ __align__(32) __bf16 sEV[32][D_IN];
    __shared__ __align__(32) __bf16 sM1[32][H_DIM];
    __shared__ __align__(32) __bf16 sM2[32][H_DIM];
    __shared__ float sDh[H_DIM];
    __shared__ float sX[H_DIM];
    __shared__ float sMu, sRs;

    stage_concat_tile(hV, hE, Eidx, i, tid, sEV);
    if (tid < H_DIM) sDh[tid] = 0.f;
    __syncthreads();

    const int col = nt * 16 + ln;
    const int rowbase = mt * 16 + hi * 8;  // C/D rows held by this lane

    // layer 1: [32,384] @ W1 -> GELU -> sM1
    {
        v8f acc; float bv = b1[col];
#pragma unroll
        for (int r = 0; r < 8; ++r) acc[r] = bv;
        const __bf16* bcol = W1t + col * D_IN;
        const __bf16* arow = &sEV[mt * 16 + ln][0];
#pragma unroll
        for (int ks = 0; ks < D_IN / 32; ++ks)
            acc = wmma_bf16(load_afrag(arow, ks * 32, hi), load_bfrag(bcol, ks * 32, hi), acc);
#pragma unroll
        for (int r = 0; r < 8; ++r) sM1[rowbase + r][col] = (__bf16)gelu_exact(acc[r]);
    }
    __syncthreads();

    // layer 2: [32,128] @ W2 -> GELU -> sM2
    {
        v8f acc; float bv = b2[col];
#pragma unroll
        for (int r = 0; r < 8; ++r) acc[r] = bv;
        const __bf16* bcol = W2t + col * H_DIM;
        const __bf16* arow = &sM1[mt * 16 + ln][0];
#pragma unroll
        for (int ks = 0; ks < H_DIM / 32; ++ks)
            acc = wmma_bf16(load_afrag(arow, ks * 32, hi), load_bfrag(bcol, ks * 32, hi), acc);
#pragma unroll
        for (int r = 0; r < 8; ++r) sM2[rowbase + r][col] = (__bf16)gelu_exact(acc[r]);
    }
    __syncthreads();

    // layer 3: [32,128] @ W3 + b3, mask_attend, partial K-sum into sDh
    {
        v8f acc; float bv = b3[col];
#pragma unroll
        for (int r = 0; r < 8; ++r) acc[r] = bv;
        const __bf16* bcol = W3t + col * H_DIM;
        const __bf16* arow = &sM2[mt * 16 + ln][0];
#pragma unroll
        for (int ks = 0; ks < H_DIM / 32; ++ks)
            acc = wmma_bf16(load_afrag(arow, ks * 32, hi), load_bfrag(bcol, ks * 32, hi), acc);
        float part = 0.f;
#pragma unroll
        for (int r = 0; r < 8; ++r) {
            int row = rowbase + r;
            float mk = (row < K_NB) ? mask_att[i * K_NB + row] : 0.f;
            part += acc[r] * mk;
        }
        atomicAdd(&sDh[col], part);   // ds_add_f32
    }
    __syncthreads();

    // residual + LayerNorm over H=128
    if (tid < H_DIM) sX[tid] = hV[(size_t)i * H_DIM + tid] + sDh[tid] * INV_SCALE;
    __syncthreads();
    if (wave == 0) {
        float s = 0.f, s2 = 0.f;
#pragma unroll
        for (int j = 0; j < 4; ++j) { float x = sX[lane + 32 * j]; s += x; s2 += x * x; }
#pragma unroll
        for (int off = 16; off > 0; off >>= 1) {
            s += __shfl_down(s, off, 32);
            s2 += __shfl_down(s2, off, 32);
        }
        if (lane == 0) {
            float mu = s * (1.0f / H_DIM);
            sMu = mu;
            sRs = rsqrtf(s2 * (1.0f / H_DIM) - mu * mu + LN_EPS);
        }
    }
    __syncthreads();
    if (tid < H_DIM)
        hV1[(size_t)i * H_DIM + tid] = (sX[tid] - sMu) * sRs * g1[tid] + be1[tid];
}

// ---------------------------------------------------------------------------
// Stage 2: position-wise FFN + residual LN + mask_V -> d_out[0 : L*H]
// 16 nodes per block (M-tile of 16). 512 threads = 16 waves.
// ---------------------------------------------------------------------------
__global__ __launch_bounds__(512) void ffn_kernel(
    const float* __restrict__ hV1, const float* __restrict__ bi_,
    const float* __restrict__ bo_, const __bf16* __restrict__ Wit,
    const __bf16* __restrict__ Wot, const float* __restrict__ g2,
    const float* __restrict__ be2, const float* __restrict__ maskV,
    float* __restrict__ hVout) {
    const int nb = blockIdx.x;        // nodes nb*16 .. nb*16+15
    const int tid = threadIdx.x;
    const int lane = tid & 31, wave = tid >> 5;
    const int hi = lane >> 4, ln = lane & 15;

    __shared__ __align__(32) __bf16 sA[16][H_DIM];
    __shared__ __align__(32) __bf16 sHid[16][F_DIM];
    __shared__ __align__(16) float sOut[16][H_DIM];

    // stage A tile: 256 chunks of 8 floats
    for (int cidx = tid; cidx < 16 * (H_DIM / 8); cidx += 512) {
        int r = cidx >> 4, cc = (cidx & 15) * 8;
        const float* src = hV1 + (size_t)(nb * 16 + r) * H_DIM + cc;
        float4 a = ((const float4*)src)[0];
        float4 b = ((const float4*)src)[1];
        bf16x8 out;
        out.v[0] = (__bf16)a.x; out.v[1] = (__bf16)a.y;
        out.v[2] = (__bf16)a.z; out.v[3] = (__bf16)a.w;
        out.v[4] = (__bf16)b.x; out.v[5] = (__bf16)b.y;
        out.v[6] = (__bf16)b.z; out.v[7] = (__bf16)b.w;
        *(bf16x8*)&sA[r][cc] = out;
    }
    __syncthreads();

    // W_in: [16,128] @ [128,512]; 32 N-tiles, each wave does 2
#pragma unroll
    for (int t = 0; t < 2; ++t) {
        int ntile = wave + t * 16;
        int col = ntile * 16 + ln;
        v8f acc; float bv = bi_[col];
#pragma unroll
        for (int r = 0; r < 8; ++r) acc[r] = bv;
        const __bf16* bcol = Wit + col * H_DIM;
        const __bf16* arow = &sA[ln][0];
#pragma unroll
        for (int ks = 0; ks < H_DIM / 32; ++ks)
            acc = wmma_bf16(load_afrag(arow, ks * 32, hi), load_bfrag(bcol, ks * 32, hi), acc);
#pragma unroll
        for (int r = 0; r < 8; ++r) sHid[hi * 8 + r][col] = (__bf16)gelu_exact(acc[r]);
    }
    __syncthreads();

    // W_out: [16,512] @ [512,128]; 8 N-tiles on waves 0..7
    if (wave < 8) {
        int col = wave * 16 + ln;
        v8f acc; float bv = bo_[col];
#pragma unroll
        for (int r = 0; r < 8; ++r) acc[r] = bv;
        const __bf16* bcol = Wot + col * F_DIM;
        const __bf16* arow = &sHid[ln][0];
#pragma unroll
        for (int ks = 0; ks < F_DIM / 32; ++ks)
            acc = wmma_bf16(load_afrag(arow, ks * 32, hi), load_bfrag(bcol, ks * 32, hi), acc);
#pragma unroll
        for (int r = 0; r < 8; ++r) sOut[hi * 8 + r][col] = acc[r];
    }
    __syncthreads();

    // per-row residual + LN + mask_V; wave w handles row w; lane covers cols lane*4..+3
    {
        int r = wave, node = nb * 16 + r;
        int c0 = lane * 4;
        float4 xr = *(const float4*)(hV1 + (size_t)node * H_DIM + c0);
        float4 xo = *(const float4*)&sOut[r][c0];
        float x[4] = { xr.x + xo.x, xr.y + xo.y, xr.z + xo.z, xr.w + xo.w };
        float s = 0.f, s2 = 0.f;
#pragma unroll
        for (int j = 0; j < 4; ++j) { s += x[j]; s2 += x[j] * x[j]; }
#pragma unroll
        for (int off = 16; off > 0; off >>= 1) {
            s += __shfl_down(s, off, 32);
            s2 += __shfl_down(s2, off, 32);
        }
        s = __shfl(s, 0, 32); s2 = __shfl(s2, 0, 32);
        float mu = s * (1.0f / H_DIM);
        float rs = rsqrtf(s2 * (1.0f / H_DIM) - mu * mu + LN_EPS);
        float mk = maskV[node];
        float4 g = *(const float4*)(g2 + c0);
        float4 bb = *(const float4*)(be2 + c0);
        float4 y;
        y.x = mk * ((x[0] - mu) * rs * g.x + bb.x);
        y.y = mk * ((x[1] - mu) * rs * g.y + bb.y);
        y.z = mk * ((x[2] - mu) * rs * g.z + bb.z);
        y.w = mk * ((x[3] - mu) * rs * g.w + bb.w);
        *(float4*)(hVout + (size_t)node * H_DIM + c0) = y;
    }
}

// ---------------------------------------------------------------------------
// Stage 3: edge MLP + residual LN with h_E -> d_out[L*H : L*H + L*K*H]
// One block per node; gathers the UPDATED h_V (hV2 = d_out head).
// ---------------------------------------------------------------------------
__global__ __launch_bounds__(512) void edge_kernel(
    const float* __restrict__ hV2, const float* __restrict__ hE,
    const int* __restrict__ Eidx,
    const float* __restrict__ b11, const float* __restrict__ b12,
    const float* __restrict__ b13,
    const __bf16* __restrict__ W11t, const __bf16* __restrict__ W12t,
    const __bf16* __restrict__ W13t,
    const float* __restrict__ g3, const float* __restrict__ be3,
    float* __restrict__ hEout) {
    const int i = blockIdx.x;
    const int tid = threadIdx.x;
    const int lane = tid & 31, wave = tid >> 5;
    const int hi = lane >> 4, ln = lane & 15;
    const int mt = wave & 1, nt = wave >> 1;

    __shared__ __align__(32) __bf16 sEV[32][D_IN];
    __shared__ __align__(32) __bf16 sM1[32][H_DIM];
    __shared__ __align__(32) __bf16 sM2[32][H_DIM];
    __shared__ __align__(16) float sMf[32][H_DIM];

    stage_concat_tile(hV2, hE, Eidx, i, tid, sEV);
    __syncthreads();

    const int col = nt * 16 + ln;
    const int rowbase = mt * 16 + hi * 8;

    {   // layer 1
        v8f acc; float bv = b11[col];
#pragma unroll
        for (int r = 0; r < 8; ++r) acc[r] = bv;
        const __bf16* bcol = W11t + col * D_IN;
        const __bf16* arow = &sEV[mt * 16 + ln][0];
#pragma unroll
        for (int ks = 0; ks < D_IN / 32; ++ks)
            acc = wmma_bf16(load_afrag(arow, ks * 32, hi), load_bfrag(bcol, ks * 32, hi), acc);
#pragma unroll
        for (int r = 0; r < 8; ++r) sM1[rowbase + r][col] = (__bf16)gelu_exact(acc[r]);
    }
    __syncthreads();
    {   // layer 2
        v8f acc; float bv = b12[col];
#pragma unroll
        for (int r = 0; r < 8; ++r) acc[r] = bv;
        const __bf16* bcol = W12t + col * H_DIM;
        const __bf16* arow = &sM1[mt * 16 + ln][0];
#pragma unroll
        for (int ks = 0; ks < H_DIM / 32; ++ks)
            acc = wmma_bf16(load_afrag(arow, ks * 32, hi), load_bfrag(bcol, ks * 32, hi), acc);
#pragma unroll
        for (int r = 0; r < 8; ++r) sM2[rowbase + r][col] = (__bf16)gelu_exact(acc[r]);
    }
    __syncthreads();
    {   // layer 3 -> f32 tile
        v8f acc; float bv = b13[col];
#pragma unroll
        for (int r = 0; r < 8; ++r) acc[r] = bv;
        const __bf16* bcol = W13t + col * H_DIM;
        const __bf16* arow = &sM2[mt * 16 + ln][0];
#pragma unroll
        for (int ks = 0; ks < H_DIM / 32; ++ks)
            acc = wmma_bf16(load_afrag(arow, ks * 32, hi), load_bfrag(bcol, ks * 32, hi), acc);
#pragma unroll
        for (int r = 0; r < 8; ++r) sMf[rowbase + r][col] = acc[r];
    }
    __syncthreads();

    // per-edge-row residual + LN; wave w handles rows w, w+16; lane covers lane*4..+3
#pragma unroll
    for (int t = 0; t < 2; ++t) {
        int r = wave + t * 16;
        if (r < K_NB) {
            const float* erow = hE + (size_t)(i * K_NB + r) * H_DIM;
            int c0 = lane * 4;
            float4 xe = *(const float4*)(erow + c0);
            float4 xm = *(const float4*)&sMf[r][c0];
            float x[4] = { xe.x + xm.x, xe.y + xm.y, xe.z + xm.z, xe.w + xm.w };
            float s = 0.f, s2 = 0.f;
#pragma unroll
            for (int j = 0; j < 4; ++j) { s += x[j]; s2 += x[j] * x[j]; }
#pragma unroll
            for (int off = 16; off > 0; off >>= 1) {
                s += __shfl_down(s, off, 32);
                s2 += __shfl_down(s2, off, 32);
            }
            s = __shfl(s, 0, 32); s2 = __shfl(s2, 0, 32);
            float mu = s * (1.0f / H_DIM);
            float rs = rsqrtf(s2 * (1.0f / H_DIM) - mu * mu + LN_EPS);
            float4 g = *(const float4*)(g3 + c0);
            float4 bb = *(const float4*)(be3 + c0);
            float4 y;
            y.x = (x[0] - mu) * rs * g.x + bb.x;
            y.y = (x[1] - mu) * rs * g.y + bb.y;
            y.z = (x[2] - mu) * rs * g.z + bb.z;
            y.w = (x[3] - mu) * rs * g.w + bb.w;
            *(float4*)(hEout + (size_t)(i * K_NB + r) * H_DIM + c0) = y;
        }
    }
}

// ---------------------------------------------------------------------------
extern "C" void kernel_launch(void* const* d_in, const int* in_sizes, int n_in,
                              void* d_out, int out_size, void* d_ws, size_t ws_size,
                              hipStream_t stream) {
    const float* hV       = (const float*)d_in[0];
    const float* hE       = (const float*)d_in[1];
    const int*   Eidx     = (const int*)d_in[2];
    const float* maskV    = (const float*)d_in[3];
    const float* maskAtt  = (const float*)d_in[4];
    const float* W1 = (const float*)d_in[5];   const float* b1 = (const float*)d_in[6];
    const float* W2 = (const float*)d_in[7];   const float* b2 = (const float*)d_in[8];
    const float* W3 = (const float*)d_in[9];   const float* b3 = (const float*)d_in[10];
    const float* W11 = (const float*)d_in[11]; const float* b11 = (const float*)d_in[12];
    const float* W12 = (const float*)d_in[13]; const float* b12 = (const float*)d_in[14];
    const float* W13 = (const float*)d_in[15]; const float* b13 = (const float*)d_in[16];
    const float* Wi = (const float*)d_in[17];  const float* bi = (const float*)d_in[18];
    const float* Wo = (const float*)d_in[19];  const float* bo = (const float*)d_in[20];
    const float* g1 = (const float*)d_in[21];  const float* be1 = (const float*)d_in[22];
    const float* g2 = (const float*)d_in[23];  const float* be2 = (const float*)d_in[24];
    const float* g3 = (const float*)d_in[25];  const float* be3 = (const float*)d_in[26];

    char* ws = (char*)d_ws;
    __bf16* W1t  = (__bf16*)(ws + OFF_W1T);
    __bf16* W2t  = (__bf16*)(ws + OFF_W2T);
    __bf16* W3t  = (__bf16*)(ws + OFF_W3T);
    __bf16* W11t = (__bf16*)(ws + OFF_W11T);
    __bf16* W12t = (__bf16*)(ws + OFF_W12T);
    __bf16* W13t = (__bf16*)(ws + OFF_W13T);
    __bf16* Wit  = (__bf16*)(ws + OFF_WIT);
    __bf16* Wot  = (__bf16*)(ws + OFF_WOT);
    float*  hV1  = (float*)(ws + OFF_HV1);

    float* hVout = (float*)d_out;                            // [L,H]
    float* hEout = (float*)d_out + (size_t)L_NODES * H_DIM;  // [L,K,H]

    // stage 0: bf16 transposed weights
    auto cvt = [&](const float* src, __bf16* dst, int din, int dout) {
        int n = din * dout;
        cvt_t_kernel<<<(n + 255) / 256, 256, 0, stream>>>(src, dst, din, dout);
    };
    cvt(W1, W1t, D_IN, H_DIM);
    cvt(W2, W2t, H_DIM, H_DIM);
    cvt(W3, W3t, H_DIM, H_DIM);
    cvt(W11, W11t, D_IN, H_DIM);
    cvt(W12, W12t, H_DIM, H_DIM);
    cvt(W13, W13t, H_DIM, H_DIM);
    cvt(Wi, Wit, H_DIM, F_DIM);
    cvt(Wo, Wot, F_DIM, H_DIM);

    // stage 1: node message + LN1 -> hV1
    node_msg_kernel<<<L_NODES, 512, 0, stream>>>(
        hV, hE, Eidx, maskAtt, b1, b2, b3, W1t, W2t, W3t, g1, be1, hV1);

    // stage 2: FFN + LN2 + mask_V -> d_out head
    ffn_kernel<<<L_NODES / 16, 512, 0, stream>>>(
        hV1, bi, bo, Wit, Wot, g2, be2, maskV, hVout);

    // stage 3: edge MLP + LN3 -> d_out tail
    edge_kernel<<<L_NODES, 512, 0, stream>>>(
        hVout, hE, Eidx, b11, b12, b13, W11t, W12t, W13t, g3, be3, hEout);
}
